// Llama4TextMoe_8332236554879
// MI455X (gfx1250) — matrix operand order repaired and checked by reading, not compiled, not run on hardware
//
#include <hip/hip_runtime.h>
#include <math.h>

// ---------------------------------------------------------------------------
// Llama4-style MoE forward for MI455X (gfx1250, wave32, WMMA bf16)
//   out = shared_swiglu(x) + routed_swiglu(score*x) ; top-1 routing
// A-tiles: GLOBAL_LOAD_ASYNC_TO_LDS_B128 (double-buffered, ASYNCcnt)
// B-tiles: fp32 -> bf16 in-register, packed ds_store_b32 transposed stage
// ---------------------------------------------------------------------------

typedef __bf16 bf16_t;
typedef __attribute__((ext_vector_type(16))) __bf16 v16bf;
typedef __attribute__((ext_vector_type(8)))  __bf16 v8bf;
typedef __attribute__((ext_vector_type(2)))  __bf16 v2bf;
typedef __attribute__((ext_vector_type(8)))  float  v8f;
typedef __attribute__((ext_vector_type(4)))  float  v4f;

#define TOK   2048   // B*S
#define HDIM  2048
#define IDIM  4096
#define NEXP  8

#define BM 64        // tile rows (tokens)
#define BN 64        // tile cols
#define BK 32        // K step (matches wmma K)
#define LDT 40       // padded LDS row stride (elements); 80B, 16B-aligned

// ---------------- small helpers ----------------
__device__ __forceinline__ v8f zero8f() {
  v8f z;
#pragma unroll
  for (int i = 0; i < 8; ++i) z[i] = 0.0f;
  return z;
}

__device__ __forceinline__ v8f wmma_bf16(v16bf a, v16bf b, v8f c) {
  // (neg_a, A, neg_b, B, c_mod, C, reuse_a, reuse_b)
  return __builtin_amdgcn_wmma_f32_16x16x32_bf16(false, a, false, b, (short)0, c,
                                                 false, false);
}

// 32-bit LDS byte offset of a __shared__ object (DS address for async VDST)
__device__ __forceinline__ unsigned lds_addr_of(const void* p) {
  return (unsigned)(size_t)(__attribute__((address_space(3))) const void*)p;
}

// one wave-level async copy: 32 lanes x 16B, global (saddr+voffset) -> LDS
__device__ __forceinline__ void async_copy16(unsigned lds_off, unsigned goff,
                                             const void* base) {
  asm volatile("global_load_async_to_lds_b128 %0, %1, %2 offset:0"
               :
               : "v"(lds_off), "v"(goff), "s"(base)
               : "memory");
}
__device__ __forceinline__ void wait_async_le1() {
  asm volatile("s_wait_asynccnt 0x1" ::: "memory");
}
__device__ __forceinline__ void wait_async_le0() {
  asm volatile("s_wait_asynccnt 0x0" ::: "memory");
}

// A fragment (16x32 bf16), ISA 7.12.2 layout:
//   lane L: m = L&15, half = L>>4; elem e -> K = (e>>3)*16 + half*8 + (e&7)
__device__ __forceinline__ v16bf frag_a(const bf16_t* As, int lane) {
  int m = lane & 15, half = lane >> 4;
  const bf16_t* row = As + m * LDT;
  v8bf lo = *(const v8bf*)(row + half * 8);
  v8bf hi = *(const v8bf*)(row + 16 + half * 8);
  v16bf a;
#pragma unroll
  for (int i = 0; i < 8; ++i) { a[i] = lo[i]; a[8 + i] = hi[i]; }
  return a;
}

// B fragment (32x16 bf16), B stored transposed in LDS: Bt[n][k].
//   lane L: n = L&15, half = L>>4; elem e -> K = (e>>3)*16 + half*8 + (e&7)
__device__ __forceinline__ v16bf frag_b(const bf16_t* Bt, int lane) {
  int n = lane & 15, half = lane >> 4;
  const bf16_t* row = Bt + n * LDT + half * 8;
  v8bf lo = *(const v8bf*)(row);
  v8bf hi = *(const v8bf*)(row + 16);
  v16bf b;
#pragma unroll
  for (int i = 0; i < 8; ++i) { b[i] = lo[i]; b[8 + i] = hi[i]; }
  return b;
}

// ---------------- tiny prep kernels ----------------
__global__ void k_cvt_x(const float* __restrict__ x, bf16_t* __restrict__ xb) {
  int i = (blockIdx.x * blockDim.x + threadIdx.x) * 4;
#pragma unroll
  for (int j = 0; j < 4; ++j) xb[i + j] = (bf16_t)x[i + j];
}

// xs[t][h] = score[t] * x[t][h] (bf16) -- routed-expert input, pre-scaled so
// the GEMM A-tile becomes a pure copy (async-LDS friendly)
__global__ void k_scale_x(const float* __restrict__ x, const float* __restrict__ score,
                          bf16_t* __restrict__ xs) {
  int i = (blockIdx.x * blockDim.x + threadIdx.x) * 4;
  float s = score[i >> 11];  // HDIM = 2048
#pragma unroll
  for (int j = 0; j < 4; ++j) xs[i + j] = (bf16_t)(x[i + j] * s);
}

// one wave per token: logits = x[t] @ router_w; top-1; sigmoid
__global__ void k_router(const float* __restrict__ x, const float* __restrict__ rw,
                         float* __restrict__ scores_out /*[E,T]*/,
                         int* __restrict__ expert, float* __restrict__ score) {
  int lane = threadIdx.x & 31;
  int t = (blockIdx.x * blockDim.x + threadIdx.x) >> 5;
  if (t >= TOK) return;
  float acc[NEXP];
#pragma unroll
  for (int e = 0; e < NEXP; ++e) acc[e] = 0.0f;
  const float* xr = x + (size_t)t * HDIM;
  for (int h = lane; h < HDIM; h += 32) {
    float xv = xr[h];
    const float* w = rw + (size_t)h * NEXP;
#pragma unroll
    for (int e = 0; e < NEXP; ++e) acc[e] += xv * w[e];
  }
#pragma unroll
  for (int off = 16; off > 0; off >>= 1) {
#pragma unroll
    for (int e = 0; e < NEXP; ++e) acc[e] += __shfl_xor(acc[e], off, 32);
  }
  if (lane == 0) {
    int best = 0;
#pragma unroll
    for (int e = 1; e < NEXP; ++e)
      if (acc[e] > acc[best]) best = e;  // strict > keeps first max (jax tie rule)
    float s = 1.0f / (1.0f + __expf(-acc[best]));
#pragma unroll
    for (int e = 0; e < NEXP; ++e)
      scores_out[(size_t)e * TOK + t] = (e == best) ? s : 0.0f;
    expert[t] = best;
    score[t] = s;
  }
}

__global__ void k_zero_counts(int* __restrict__ counts) {
  if (threadIdx.x < NEXP) counts[threadIdx.x] = 0;
}
__global__ void k_scatter(const int* __restrict__ expert, int* __restrict__ counts,
                          int* __restrict__ slot) {
  int t = blockIdx.x * blockDim.x + threadIdx.x;
  if (t < TOK) slot[t] = atomicAdd(&counts[expert[t]], 1);
}
__global__ void k_offsets(const int* __restrict__ counts, int* __restrict__ offsets) {
  if (threadIdx.x == 0) {
    int a = 0;
    for (int e = 0; e < NEXP; ++e) { offsets[e] = a; a += counts[e]; }
  }
}
__global__ void k_build_rows(const int* __restrict__ expert, const int* __restrict__ slot,
                             const int* __restrict__ offsets, int* __restrict__ tokrow) {
  int t = blockIdx.x * blockDim.x + threadIdx.x;
  if (t < TOK) tokrow[offsets[expert[t]] + slot[t]] = t;
}

// ---------------- fused SwiGLU up: mid = silu(A@Wg) * (A@Wu) ----------------
// grid: (IDIM/BN, max m-tiles, NEXP or 1). 256 thr = 8 waves (4 Mx2 N groups).
template <bool ROUTED>
__global__ __launch_bounds__(256) void k_mlp_up(
    const bf16_t* __restrict__ xin,  // shared: xb ; routed: xs (pre-scaled)
    const float* __restrict__ gup, const float* __restrict__ sgw,
    const float* __restrict__ suw, bf16_t* __restrict__ mid,
    const int* __restrict__ counts, const int* __restrict__ offsets,
    const int* __restrict__ tokrow) {
  int e = blockIdx.z;
  int cnt = ROUTED ? counts[e] : TOK;
  int row0 = ROUTED ? offsets[e] : 0;
  int m_base = blockIdx.y * BM;
  if (m_base >= cnt) return;
  int n_base = blockIdx.x * BN;

  const float* Bg;
  const float* Bu;
  long sB;
  if (ROUTED) {
    Bg = gup + (size_t)e * HDIM * (2 * IDIM);
    Bu = Bg + IDIM;
    sB = 2 * IDIM;
  } else {
    Bg = sgw;
    Bu = suw;
    sB = IDIM;
  }

  __shared__ __align__(16) bf16_t As0[BM * LDT];
  __shared__ __align__(16) bf16_t As1[BM * LDT];
  __shared__ __align__(16) bf16_t Bgt[BN * LDT];
  __shared__ __align__(16) bf16_t But[BN * LDT];

  int tid = threadIdx.x;
  // ---- A staging (async): thread -> (row 0..63, 16B k-group) ----
  int ar = tid >> 2, ak0 = (tid & 3) * 8;
  int arow = m_base + ar;
  bool avalid = arow < cnt;
  int atok = ROUTED ? tokrow[row0 + (avalid ? arow : 0)] : (avalid ? arow : 0);
  unsigned agoff = (unsigned)(((size_t)atok * HDIM + ak0) * sizeof(bf16_t));
  unsigned al0 = lds_addr_of(As0 + ar * LDT + ak0);
  unsigned al1 = lds_addr_of(As1 + ar * LDT + ak0);

  // ---- B staging: thread -> (k-pair 0..15, 4-col n group) ----
  int kp = (tid >> 4) * 2;      // k0 in {0,2,...,30}
  int bn0 = (tid & 15) * 4;     // n0 in {0,4,...,60}
  const float* gp = Bg + (size_t)kp * sB + n_base + bn0;
  const float* up = Bu + (size_t)kp * sB + n_base + bn0;

  float g0r[4], g1r[4], u0r[4], u1r[4];
  auto fetchB = [&](int kt) {
    size_t rb = (size_t)kt * BK * sB;
    v4f a = *(const v4f*)(gp + rb);
    v4f b = *(const v4f*)(gp + rb + sB);
    v4f c = *(const v4f*)(up + rb);
    v4f d = *(const v4f*)(up + rb + sB);
#pragma unroll
    for (int j = 0; j < 4; ++j) { g0r[j] = a[j]; g1r[j] = b[j]; u0r[j] = c[j]; u1r[j] = d[j]; }
  };
  auto stageB = [&]() {
#pragma unroll
    for (int j = 0; j < 4; ++j) {
      v2bf tg; tg[0] = (bf16_t)g0r[j]; tg[1] = (bf16_t)g1r[j];
      *(v2bf*)(Bgt + (bn0 + j) * LDT + kp) = tg;   // packed ds_store_b32
      v2bf tu; tu[0] = (bf16_t)u0r[j]; tu[1] = (bf16_t)u1r[j];
      *(v2bf*)(But + (bn0 + j) * LDT + kp) = tu;
    }
  };

  int lane = tid & 31, wave = tid >> 5;
  int wm = wave & 3, wn = wave >> 2;
  const bf16_t* g0B = Bgt + (wn * 32) * LDT;
  const bf16_t* g1B = Bgt + (wn * 32 + 16) * LDT;
  const bf16_t* u0B = But + (wn * 32) * LDT;
  const bf16_t* u1B = But + (wn * 32 + 16) * LDT;

  v8f cg0 = zero8f(), cg1 = zero8f(), cu0 = zero8f(), cu1 = zero8f();

  const int KT = HDIM / BK;
  async_copy16(al0, agoff, xin);  // A tile kt=0 -> buf0
  fetchB(0);
  for (int kt = 0; kt < KT; ++kt) {
    __syncthreads();              // prev compute done reading LDS
    stageB();
    if (kt + 1 < KT) {            // A tile kt+1 -> other buffer (overlaps compute)
      async_copy16(((kt + 1) & 1) ? al1 : al0,
                   agoff + (unsigned)(kt + 1) * (BK * sizeof(bf16_t)), xin);
      wait_async_le1();           // kt's async copy landed (in-order completion)
    } else {
      wait_async_le0();
    }
    __syncthreads();              // all A + B LDS writes visible
    if (kt + 1 < KT) fetchB(kt + 1);  // global loads overlap compute below
    const bf16_t* aB = ((kt & 1) ? As1 : As0) + (wm * 16) * LDT;
    v16bf a = frag_a(aB, lane);
    v16bf b;
    b = frag_b(g0B, lane); cg0 = wmma_bf16(a, b, cg0);
    b = frag_b(g1B, lane); cg1 = wmma_bf16(a, b, cg1);
    b = frag_b(u0B, lane); cu0 = wmma_bf16(a, b, cu0);
    b = frag_b(u1B, lane); cu1 = wmma_bf16(a, b, cu1);
  }

  // epilogue: silu(gate)*up -> bf16 mid[row0+row][col]
  int half = lane >> 4, nl = lane & 15;
#pragma unroll
  for (int s = 0; s < 2; ++s) {
    v8f g = s ? cg1 : cg0;
    v8f u = s ? cu1 : cu0;
    int col = n_base + wn * 32 + s * 16 + nl;
#pragma unroll
    for (int r = 0; r < 8; ++r) {
      int row = m_base + wm * 16 + half * 8 + r;
      if (row < cnt) {
        float gv = g[r];
        float val = (gv / (1.0f + __expf(-gv))) * u[r];
        mid[(size_t)(row0 + row) * IDIM + col] = (bf16_t)val;
      }
    }
  }
}

// ---------------- down projection: out[token] (=|+=) mid @ Wd ----------------
template <bool ROUTED>
__global__ __launch_bounds__(256) void k_mlp_down(
    const bf16_t* __restrict__ mid, const float* __restrict__ dw,
    float* __restrict__ out, const int* __restrict__ counts,
    const int* __restrict__ offsets, const int* __restrict__ tokrow) {
  int e = blockIdx.z;
  int cnt = ROUTED ? counts[e] : TOK;
  int row0 = ROUTED ? offsets[e] : 0;
  int m_base = blockIdx.y * BM;
  if (m_base >= cnt) return;
  int n_base = blockIdx.x * BN;
  const float* B = ROUTED ? dw + (size_t)e * IDIM * HDIM : dw;
  const long sB = HDIM;

  __shared__ __align__(16) bf16_t As0[BM * LDT];
  __shared__ __align__(16) bf16_t As1[BM * LDT];
  __shared__ __align__(16) bf16_t Bt[BN * LDT];

  int tid = threadIdx.x;
  int ar = tid >> 2, ak0 = (tid & 3) * 8;
  int arow = m_base + ar;
  bool avalid = arow < cnt;
  size_t amrow = (size_t)(row0 + (avalid ? arow : 0));
  unsigned agoff = (unsigned)((amrow * IDIM + ak0) * sizeof(bf16_t));
  unsigned al0 = lds_addr_of(As0 + ar * LDT + ak0);
  unsigned al1 = lds_addr_of(As1 + ar * LDT + ak0);

  int kp = (tid >> 4) * 2;
  int bn0 = (tid & 15) * 4;
  const float* bp = B + (size_t)kp * sB + n_base + bn0;

  float b0r[4], b1r[4];
  auto fetchB = [&](int kt) {
    size_t rb = (size_t)kt * BK * sB;
    v4f a = *(const v4f*)(bp + rb);
    v4f b = *(const v4f*)(bp + rb + sB);
#pragma unroll
    for (int j = 0; j < 4; ++j) { b0r[j] = a[j]; b1r[j] = b[j]; }
  };
  auto stageB = [&]() {
#pragma unroll
    for (int j = 0; j < 4; ++j) {
      v2bf t; t[0] = (bf16_t)b0r[j]; t[1] = (bf16_t)b1r[j];
      *(v2bf*)(Bt + (bn0 + j) * LDT + kp) = t;
    }
  };

  int lane = tid & 31, wave = tid >> 5;
  int wm = wave & 3, wn = wave >> 2;
  const bf16_t* b0B = Bt + (wn * 32) * LDT;
  const bf16_t* b1B = Bt + (wn * 32 + 16) * LDT;

  v8f c0 = zero8f(), c1 = zero8f();

  const int KT = IDIM / BK;
  async_copy16(al0, agoff, mid);
  fetchB(0);
  for (int kt = 0; kt < KT; ++kt) {
    __syncthreads();
    stageB();
    if (kt + 1 < KT) {
      async_copy16(((kt + 1) & 1) ? al1 : al0,
                   agoff + (unsigned)(kt + 1) * (BK * sizeof(bf16_t)), mid);
      wait_async_le1();
    } else {
      wait_async_le0();
    }
    __syncthreads();
    if (kt + 1 < KT) fetchB(kt + 1);
    const bf16_t* aB = ((kt & 1) ? As1 : As0) + (wm * 16) * LDT;
    v16bf a = frag_a(aB, lane);
    v16bf b;
    b = frag_b(b0B, lane); c0 = wmma_bf16(a, b, c0);
    b = frag_b(b1B, lane); c1 = wmma_bf16(a, b, c1);
  }

  int half = lane >> 4, nl = lane & 15;
#pragma unroll
  for (int s = 0; s < 2; ++s) {
    v8f c = s ? c1 : c0;
    int col = n_base + wn * 32 + s * 16 + nl;
#pragma unroll
    for (int r = 0; r < 8; ++r) {
      int row = m_base + wm * 16 + half * 8 + r;
      if (row < cnt) {
        int tok = ROUTED ? tokrow[row0 + row] : row;
        float v = c[r];
        float* dst = out + (size_t)tok * HDIM + col;
        if (ROUTED)
          *dst += v;  // token rows exclusive per expert (top-1); ordered after shared
        else
          *dst = v;
      }
    }
  }
}

// ---------------------------------------------------------------------------
extern "C" void kernel_launch(void* const* d_in, const int* in_sizes, int n_in,
                              void* d_out, int out_size, void* d_ws, size_t ws_size,
                              hipStream_t stream) {
  (void)in_sizes; (void)n_in; (void)out_size; (void)ws_size;
  const float* x   = (const float*)d_in[0];  // [B,S,H]
  const float* rw  = (const float*)d_in[1];  // [H,E]
  const float* gup = (const float*)d_in[2];  // [E,H,2I]
  const float* dwp = (const float*)d_in[3];  // [E,I,H]
  const float* sgw = (const float*)d_in[4];  // [H,I]
  const float* suw = (const float*)d_in[5];  // [H,I]
  const float* sdw = (const float*)d_in[6];  // [I,H]

  float* out = (float*)d_out;                 // [T*H]
  float* rsc = out + (size_t)TOK * HDIM;      // [E*T]

  char* ws = (char*)d_ws;
  size_t off = 0;
  auto walloc = [&](size_t bytes) -> void* {
    void* p = ws + off;
    off = (off + bytes + 255) & ~(size_t)255;
    return p;
  };
  bf16_t* xb     = (bf16_t*)walloc((size_t)TOK * HDIM * 2);
  bf16_t* xs     = (bf16_t*)walloc((size_t)TOK * HDIM * 2);
  bf16_t* mid_sh = (bf16_t*)walloc((size_t)TOK * IDIM * 2);
  bf16_t* mid_rt = (bf16_t*)walloc((size_t)TOK * IDIM * 2);
  int*    expert = (int*)walloc(TOK * 4);
  float*  score  = (float*)walloc(TOK * 4);
  int*    slot   = (int*)walloc(TOK * 4);
  int*    counts = (int*)walloc(NEXP * 4);
  int*    offs   = (int*)walloc(NEXP * 4);
  int*    tokrow = (int*)walloc(TOK * 4);

  k_cvt_x<<<(TOK * HDIM / 4 + 255) / 256, 256, 0, stream>>>(x, xb);
  k_router<<<TOK / 8, 256, 0, stream>>>(x, rw, rsc, expert, score);
  k_scale_x<<<(TOK * HDIM / 4 + 255) / 256, 256, 0, stream>>>(x, score, xs);
  k_zero_counts<<<1, 32, 0, stream>>>(counts);
  k_scatter<<<TOK / 256, 256, 0, stream>>>(expert, counts, slot);
  k_offsets<<<1, 32, 0, stream>>>(counts, offs);
  k_build_rows<<<TOK / 256, 256, 0, stream>>>(expert, slot, offs, tokrow);

  dim3 gu_sh(IDIM / BN, TOK / BM, 1);
  k_mlp_up<false><<<gu_sh, 256, 0, stream>>>(xb, nullptr, sgw, suw, mid_sh,
                                             counts, offs, tokrow);
  dim3 gu_rt(IDIM / BN, TOK / BM, NEXP);
  k_mlp_up<true><<<gu_rt, 256, 0, stream>>>(xs, gup, nullptr, nullptr, mid_rt,
                                            counts, offs, tokrow);
  dim3 gd_sh(HDIM / BN, TOK / BM, 1);
  k_mlp_down<false><<<gd_sh, 256, 0, stream>>>(mid_sh, sdw, out, counts, offs, tokrow);
  dim3 gd_rt(HDIM / BN, TOK / BM, NEXP);
  k_mlp_down<true><<<gd_rt, 256, 0, stream>>>(mid_rt, dwp, out, counts, offs, tokrow);
}
